// LSTM_73512660239038
// MI455X (gfx1250) — compile-verified, matrix operand order checked by reading, not run on hardware
//
#include <hip/hip_runtime.h>
#include <hip/hip_bf16.h>

// ---------------------------------------------------------------------------
// LSTM forward on gfx1250 via v_wmma_f32_16x16x32_bf16.
// H = B = 1024, T = 64.
// h and c are kept TRANSPOSED [B, H] (n-major) so all tile staging and
// epilogue I/O is 16-byte vectorized. h double-buffered bf16, c fp32 in-place.
//   step t:  z[4H,B] = Wh_bf16[4H,H] @ h_bf16[H,B]  (WMMA, fp32 acc)
//            z += wx[4H] * x[:,t]  (rank-1, epilogue)
//            bias broadcast over COLUMNS (reference quirk): + b[n]
//   final:   out[n,m] = sum_k w_p[m,k] * h[k,n] + b_p[m]
// ---------------------------------------------------------------------------

typedef __attribute__((ext_vector_type(16))) __bf16 v16bf;
typedef __attribute__((ext_vector_type(8)))  float  v8f;

#define HID   1024
#define BATCH 1024
#define SEQT  64
#define KCH   32                    // K-chunk (WMMA K)
#define NCHUNK (HID / KCH)          // 32 chunks

union FragU { uint4 u2[2]; v16bf v; };
union PackU { unsigned short s[8]; uint4 u; };

__device__ __forceinline__ unsigned short f32_to_bf16(float x) {
    unsigned u = __builtin_bit_cast(unsigned, x);
    unsigned r = u + 0x7FFFu + ((u >> 16) & 1u);
    return (unsigned short)(r >> 16);
}

// Fragment (16-bit A 16x32 / B 32x16 mirrored): lane&15 = row(A)/col(B);
// lanes>=16 take K+8 / K+24 halves. p -> K=0 of this lane's 32-element row.
__device__ __forceinline__ v16bf ld_frag(const unsigned short* p) {
    FragU f;
    f.u2[0] = *reinterpret_cast<const uint4*>(p);        // K 0..7
    f.u2[1] = *reinterpret_cast<const uint4*>(p + 16);   // K 16..23
    return f.v;
}

// ---------------------------------------------------------------------------
// Weight prep (runs every call; trivially bandwidth-bound, fits in L2).
// ---------------------------------------------------------------------------
__global__ void cvt_gates(const float* __restrict__ wg, const float* __restrict__ wi,
                          const float* __restrict__ wf, const float* __restrict__ wo,
                          unsigned short* __restrict__ Wh, float* __restrict__ wx) {
    const int H = HID;
    long long i = (long long)blockIdx.x * blockDim.x + threadIdx.x;
    const long long total = 4LL * H * (H + 1);
    if (i >= total) return;
    int g   = (int)(i / ((long long)H * (H + 1)));
    int rem = (int)(i % ((long long)H * (H + 1)));
    int r   = rem / (H + 1);
    int col = rem % (H + 1);
    const float* W = (g == 0) ? wg : (g == 1) ? wi : (g == 2) ? wf : wo;
    float v = W[rem];
    if (col == 0) wx[g * H + r] = v;
    else Wh[((long long)(g * H + r)) * H + (col - 1)] = f32_to_bf16(v);
}

__global__ void cvt_wp(const float* __restrict__ wp, unsigned short* __restrict__ wpb) {
    int i = blockIdx.x * blockDim.x + threadIdx.x;
    if (i < HID * HID) wpb[i] = f32_to_bf16(wp[i]);
}

__global__ void zero_u32(unsigned int* __restrict__ p, int n) {
    int i = blockIdx.x * blockDim.x + threadIdx.x;
    if (i < n) p[i] = 0u;
}

// ---------------------------------------------------------------------------
// One LSTM timestep. Grid (H/64, B/64), 256 threads (8 waves).
// WG tile: 64 rows (per gate, all 4 gates) x 64 batch cols.
// Wave wv: N-subtile (wv&3), M-subtile pair (wv>>2) -> acc[4 gates][2 mt].
// Double-buffered LDS, one barrier per K-chunk.
// ---------------------------------------------------------------------------
__global__ void __launch_bounds__(256)
lstm_step(const unsigned short* __restrict__ Wh,   // [4H, H] bf16
          const float* __restrict__ wx,            // [4H]
          const unsigned short* __restrict__ hT_in,// [B, H] bf16
          unsigned short* __restrict__ hT_out,     // [B, H] bf16
          float* __restrict__ cT,                  // [B, H] f32 (in-place)
          const float* __restrict__ x,             // [B, T]
          const float* __restrict__ b_g, const float* __restrict__ b_i,
          const float* __restrict__ b_f, const float* __restrict__ b_o,
          int t) {
    const int H = HID;
    __shared__ __align__(16) unsigned short As[2][4 * 64 * KCH]; // 2 x 16 KB
    __shared__ __align__(16) unsigned short Bs[2][64 * KCH];     // 2 x  4 KB

    const int tid   = threadIdx.x;
    const int lane  = tid & 31;
    const int wv    = tid >> 5;
    const int row15 = lane & 15;
    const int khalf = (lane >> 4) * 8;
    const int m0 = blockIdx.x * 64;
    const int n0 = blockIdx.y * 64;

    v8f acc[4][2];
#pragma unroll
    for (int g = 0; g < 4; ++g)
#pragma unroll
        for (int mt = 0; mt < 2; ++mt) acc[g][mt] = v8f{};

    auto fill = [&](int buf, int kc) {
        // A: 4 gates x 64 rows x 32 K -> 1024 uint4 (4 per thread)
#pragma unroll
        for (int it = 0; it < 4; ++it) {
            int j4 = tid + it * 256;
            int g = j4 >> 8, m = (j4 >> 2) & 63, quad = j4 & 3;
            reinterpret_cast<uint4*>(As[buf])[j4] =
                *reinterpret_cast<const uint4*>(Wh + (g * H + m0 + m) * H + kc + quad * 8);
        }
        // B: 64 cols x 32 K from hT rows (contiguous) -> 256 uint4 (1/thread)
        {
            int n = tid >> 2, quad = tid & 3;
            reinterpret_cast<uint4*>(Bs[buf])[tid] =
                *reinterpret_cast<const uint4*>(hT_in + (n0 + n) * H + kc + quad * 8);
        }
    };

    fill(0, 0);
    const int nsub = wv & 3;
    const int mthi = (wv >> 2) * 2;

    for (int i = 0; i < NCHUNK; ++i) {
        __syncthreads();
        int buf = i & 1;
        if (i + 1 < NCHUNK) fill(buf ^ 1, (i + 1) * KCH);
        if (i + 2 < NCHUNK) {   // global_prefetch_b8 the K+2 chunks
            __builtin_prefetch(Wh + (m0) * H + (i + 2) * KCH, 0, 1);
            __builtin_prefetch(hT_in + (n0) * H + (i + 2) * KCH, 0, 1);
        }
        v16bf bfrag = ld_frag(&Bs[buf][(nsub * 16 + row15) * KCH + khalf]);
#pragma unroll
        for (int g = 0; g < 4; ++g) {
#pragma unroll
            for (int mt = 0; mt < 2; ++mt) {
                int mrow = (mthi + mt) * 16 + row15;
                v16bf afrag = ld_frag(&As[buf][(g * 64 + mrow) * KCH + khalf]);
                acc[g][mt] = __builtin_amdgcn_wmma_f32_16x16x32_bf16(
                    false, afrag, false, bfrag, (short)0, acc[g][mt], false, false);
            }
        }
    }

    // Epilogue. C layout: VGPR r -> M = r + (lane>=16 ? 8 : 0), N = lane&15.
    const int n   = n0 + nsub * 16 + row15;
    const int mhi = (lane >> 4) * 8;
    const float xv  = x[n * SEQT + t];
    const float bgv = b_g[n], biv = b_i[n], bfv = b_f[n], bov = b_o[n];

#pragma unroll
    for (int mt = 0; mt < 2; ++mt) {
        const int base_m = m0 + (mthi + mt) * 16 + mhi;   // 8 consecutive m
        float4 wgl = *reinterpret_cast<const float4*>(wx + base_m);
        float4 wgh = *reinterpret_cast<const float4*>(wx + base_m + 4);
        float4 wil = *reinterpret_cast<const float4*>(wx + HID + base_m);
        float4 wih = *reinterpret_cast<const float4*>(wx + HID + base_m + 4);
        float4 wfl = *reinterpret_cast<const float4*>(wx + 2 * HID + base_m);
        float4 wfh = *reinterpret_cast<const float4*>(wx + 2 * HID + base_m + 4);
        float4 wol = *reinterpret_cast<const float4*>(wx + 3 * HID + base_m);
        float4 woh = *reinterpret_cast<const float4*>(wx + 3 * HID + base_m + 4);
        float wxg[8] = {wgl.x, wgl.y, wgl.z, wgl.w, wgh.x, wgh.y, wgh.z, wgh.w};
        float wxi[8] = {wil.x, wil.y, wil.z, wil.w, wih.x, wih.y, wih.z, wih.w};
        float wxf[8] = {wfl.x, wfl.y, wfl.z, wfl.w, wfh.x, wfh.y, wfh.z, wfh.w};
        float wxo[8] = {wol.x, wol.y, wol.z, wol.w, woh.x, woh.y, woh.z, woh.w};

        float* crow = cT + (long long)n * H + base_m;
        float4 cl = *reinterpret_cast<const float4*>(crow);
        float4 ch = *reinterpret_cast<const float4*>(crow + 4);
        float cold[8] = {cl.x, cl.y, cl.z, cl.w, ch.x, ch.y, ch.z, ch.w};

        float c2[8];
        PackU hp;
#pragma unroll
        for (int r = 0; r < 8; ++r) {
            float zg = acc[0][mt][r] + wxg[r] * xv + bgv;
            float zi = acc[1][mt][r] + wxi[r] * xv + biv;
            float zf = acc[2][mt][r] + wxf[r] * xv + bfv;
            float zo = acc[3][mt][r] + wxo[r] * xv + bov;
            float gg = tanhf(zg);
            float ii = 1.f / (1.f + __expf(-zi));
            float ff = 1.f / (1.f + __expf(-zf));
            float oo = 1.f / (1.f + __expf(-zo));
            c2[r] = gg * ii + cold[r] * ff;
            hp.s[r] = f32_to_bf16(tanhf(c2[r]) * oo);
        }
        *reinterpret_cast<float4*>(crow)     = make_float4(c2[0], c2[1], c2[2], c2[3]);
        *reinterpret_cast<float4*>(crow + 4) = make_float4(c2[4], c2[5], c2[6], c2[7]);
        *reinterpret_cast<uint4*>(hT_out + (long long)n * H + base_m) = hp.u;
    }
}

// ---------------------------------------------------------------------------
// Final projection: out[n, m] = sum_k w_p[m,k] * h[k,n] + b_p[m]
// Grid (H/64, B/64), same tiling; single "gate".
// ---------------------------------------------------------------------------
__global__ void __launch_bounds__(256)
proj_kernel(const unsigned short* __restrict__ Wp,    // [H, H] bf16
            const unsigned short* __restrict__ hT_in, // [B, H] bf16
            const float* __restrict__ b_p,
            float* __restrict__ out) {                // [B, H]
    const int H = HID;
    __shared__ __align__(16) unsigned short As[2][64 * KCH]; // 2 x 4 KB
    __shared__ __align__(16) unsigned short Bs[2][64 * KCH]; // 2 x 4 KB

    const int tid   = threadIdx.x;
    const int lane  = tid & 31;
    const int wv    = tid >> 5;
    const int row15 = lane & 15;
    const int khalf = (lane >> 4) * 8;
    const int m0 = blockIdx.x * 64;
    const int n0 = blockIdx.y * 64;

    v8f acc[2];
#pragma unroll
    for (int mt = 0; mt < 2; ++mt) acc[mt] = v8f{};

    auto fill = [&](int buf, int kc) {
        {   // A: 64 rows x 32 K -> 256 uint4
            int m = tid >> 2, quad = tid & 3;
            reinterpret_cast<uint4*>(As[buf])[tid] =
                *reinterpret_cast<const uint4*>(Wp + (m0 + m) * H + kc + quad * 8);
        }
        {   // B: 64 cols x 32 K -> 256 uint4
            int n = tid >> 2, quad = tid & 3;
            reinterpret_cast<uint4*>(Bs[buf])[tid] =
                *reinterpret_cast<const uint4*>(hT_in + (n0 + n) * H + kc + quad * 8);
        }
    };

    fill(0, 0);
    const int nsub = wv & 3;
    const int mthi = (wv >> 2) * 2;

    for (int i = 0; i < NCHUNK; ++i) {
        __syncthreads();
        int buf = i & 1;
        if (i + 1 < NCHUNK) fill(buf ^ 1, (i + 1) * KCH);
        v16bf bfrag = ld_frag(&Bs[buf][(nsub * 16 + row15) * KCH + khalf]);
#pragma unroll
        for (int mt = 0; mt < 2; ++mt) {
            int mrow = (mthi + mt) * 16 + row15;
            v16bf afrag = ld_frag(&As[buf][mrow * KCH + khalf]);
            acc[mt] = __builtin_amdgcn_wmma_f32_16x16x32_bf16(
                false, afrag, false, bfrag, (short)0, acc[mt], false, false);
        }
    }

    const int n   = n0 + nsub * 16 + row15;
    const int mhi = (lane >> 4) * 8;
#pragma unroll
    for (int mt = 0; mt < 2; ++mt) {
        const int base_m = m0 + (mthi + mt) * 16 + mhi;
        float4 bl = *reinterpret_cast<const float4*>(b_p + base_m);
        float4 bh = *reinterpret_cast<const float4*>(b_p + base_m + 4);
        float* orow = out + (long long)n * H + base_m;
        *reinterpret_cast<float4*>(orow) =
            make_float4(acc[mt][0] + bl.x, acc[mt][1] + bl.y,
                        acc[mt][2] + bl.z, acc[mt][3] + bl.w);
        *reinterpret_cast<float4*>(orow + 4) =
            make_float4(acc[mt][4] + bh.x, acc[mt][5] + bh.y,
                        acc[mt][6] + bh.z, acc[mt][7] + bh.w);
    }
}

// ---------------------------------------------------------------------------
extern "C" void kernel_launch(void* const* d_in, const int* in_sizes, int n_in,
                              void* d_out, int out_size, void* d_ws, size_t ws_size,
                              hipStream_t stream) {
    (void)in_sizes; (void)n_in; (void)out_size; (void)ws_size;
    const int H = HID, Bt = BATCH;

    const float* x   = (const float*)d_in[0];
    const float* w_g = (const float*)d_in[1];
    const float* b_g = (const float*)d_in[2];
    const float* w_i = (const float*)d_in[3];
    const float* b_i = (const float*)d_in[4];
    const float* w_f = (const float*)d_in[5];
    const float* b_f = (const float*)d_in[6];
    const float* w_o = (const float*)d_in[7];
    const float* b_o = (const float*)d_in[8];
    const float* w_p = (const float*)d_in[9];
    const float* b_p = (const float*)d_in[10];
    float* out = (float*)d_out;

    char* ws = (char*)d_ws;
    size_t off = 0;
    auto carve = [&](size_t bytes) {
        char* p = ws + off;
        off += (bytes + 255) & ~(size_t)255;
        return p;
    };
    unsigned short* Wh  = (unsigned short*)carve((size_t)4 * H * H * 2); // 8 MB
    unsigned short* Wpb = (unsigned short*)carve((size_t)H * H * 2);     // 2 MB
    float*          wx  = (float*)carve((size_t)4 * H * 4);
    unsigned short* h0  = (unsigned short*)carve((size_t)Bt * H * 2);    // 2 MB [B,H]
    unsigned short* h1  = (unsigned short*)carve((size_t)Bt * H * 2);    // 2 MB [B,H]
    float*          c   = (float*)carve((size_t)Bt * H * 4);             // 4 MB [B,H]

    {   // weight prep
        long long total = 4LL * H * (H + 1);
        cvt_gates<<<(int)((total + 255) / 256), 256, 0, stream>>>(w_g, w_i, w_f, w_o, Wh, wx);
        cvt_wp<<<(H * H + 255) / 256, 256, 0, stream>>>(w_p, Wpb);
    }
    {   // zero h0 (read at t=0) and c
        int nh = Bt * H / 2, nc = Bt * H;
        zero_u32<<<(nh + 255) / 256, 256, 0, stream>>>((unsigned int*)h0, nh);
        zero_u32<<<(nc + 255) / 256, 256, 0, stream>>>((unsigned int*)c, nc);
    }
    unsigned short* hbuf[2] = {h0, h1};
    dim3 grid(H / 64, Bt / 64);
    for (int t = 0; t < SEQT; ++t) {
        lstm_step<<<grid, 256, 0, stream>>>(Wh, wx, hbuf[t & 1], hbuf[(t + 1) & 1],
                                            c, x, b_g, b_i, b_f, b_o, t);
    }
    // After t=63, final h lives in hbuf[0].
    proj_kernel<<<grid, 256, 0, stream>>>(Wpb, hbuf[0], b_p, out);
}
